// POS_tag_net_36455682408737
// MI455X (gfx1250) — matrix operand order repaired and checked by reading, not compiled, not run on hardware
//
#include <hip/hip_runtime.h>
#include <hip/hip_bf16.h>
#include <math.h>

// ---------------------------------------------------------------------------
// POS tagger forward pass for MI455X (gfx1250, wave32, WMMA).
//   B=128, S=512, W=20, V=50000, C=100, T=50, DE=128, DC=32, OC=64, H=256, K=3
// LSTM scan axis = B (torch batch_first=False quirk), "batch" = S.
// All GEMMs run through v_wmma_f32_16x16x32_f16 (f16 in, f32 accumulate).
// M-tile = 32 rows/block: every B (weight) fragment feeds two WMMAs, halving
// L2 weight traffic vs a 16-row tile (the binding constraint for the small
// per-step GEMMs).
// ---------------------------------------------------------------------------

#define Bn   128
#define Sn   512
#define Wn   20
#define DEn  128
#define DCn  32
#define OCn  64
#define Hn   256
#define Gn   1024      // 4*H
#define Dxn  192       // DE + OC
#define Tn   50
#define H2n  512       // 2*H
#define NBS  (Bn*Sn)   // 65536

typedef __attribute__((ext_vector_type(16))) _Float16 v16h;
typedef __attribute__((ext_vector_type(8)))  _Float16 v8h;
typedef __attribute__((ext_vector_type(8)))  float    v8f;

union F16x16 { v16h v; v8h h[2]; _Float16 e[16]; };
union F32x8  { v8f  v; float e[8]; };

// ------------------------------ small prep kernels -------------------------

__global__ __launch_bounds__(256) void cvt_f32_f16_kernel(
    const float* __restrict__ s, _Float16* __restrict__ d, int n) {
  int i = blockIdx.x * 256 + threadIdx.x;
  if (i < n) d[i] = (_Float16)s[i];
}

__global__ __launch_bounds__(256) void zero_f16_kernel(_Float16* d, int n) {
  int i = blockIdx.x * 256 + threadIdx.x;
  if (i < n) d[i] = (_Float16)0.0f;
}

__global__ __launch_bounds__(256) void zero_f32_kernel(float* d, int n) {
  int i = blockIdx.x * 256 + threadIdx.x;
  if (i < n) d[i] = 0.0f;
}

// bias[dir][g] = b_ih[dir][g] + b_hh[dir][g]
__global__ __launch_bounds__(256) void bias_sum_kernel(
    const float* __restrict__ bihf, const float* __restrict__ bhhf,
    const float* __restrict__ bihb, const float* __restrict__ bhhb,
    float* __restrict__ bias) {
  int i = blockIdx.x * 256 + threadIdx.x;
  if (i < Gn)            bias[i] = bihf[i] + bhhf[i];
  else if (i < 2 * Gn)   bias[i] = bihb[i - Gn] + bhhb[i - Gn];
}

// ------------------------- embeddings & char CNN ---------------------------

__global__ __launch_bounds__(128) void embed_word_kernel(
    const int* __restrict__ words, const int* __restrict__ wlens,
    const float* __restrict__ wemb, _Float16* __restrict__ xin) {
  int bs = blockIdx.x;                 // 0..NBS-1
  int b  = bs >> 9, s = bs & 511;
  int w  = words[bs];
  bool valid = s < wlens[b];
  float v = valid ? wemb[(size_t)w * DEn + threadIdx.x] : 0.0f;
  xin[(size_t)bs * Dxn + threadIdx.x] = (_Float16)v;
}

__global__ __launch_bounds__(64) void char_cnn_kernel(
    const int* __restrict__ chars, const int* __restrict__ wlens,
    const int* __restrict__ clens, const float* __restrict__ cemb,
    const float* __restrict__ cw, const float* __restrict__ cb,
    _Float16* __restrict__ xin) {
  __shared__ float ce[Wn + 2][DCn];    // zero halo rows 0 and 21
  int bs = blockIdx.x;
  int b  = bs >> 9, s = bs & 511;
  int tid = threadIdx.x;
  bool srow = s < wlens[b];
  int clen = clens[bs];

  for (int idx = tid; idx < (Wn + 2) * DCn; idx += 64) {
    int w = idx >> 5, ic = idx & 31;
    float v = 0.0f;
    if (w >= 1 && w <= Wn) {
      int wp = w - 1;
      if (srow && wp < clen) {
        int ch = chars[bs * Wn + wp];
        v = cemb[ch * DCn + ic];
      }
    }
    ce[w][ic] = v;
  }
  __syncthreads();

  int oc = tid;                        // 64 threads == 64 output channels
  float wr[DCn * 3];
#pragma unroll
  for (int j = 0; j < DCn * 3; ++j) wr[j] = cw[oc * DCn * 3 + j];

  float bb = cb[oc];
  float m = 0.0f;                      // max(relu(x)) >= 0
#pragma unroll 1
  for (int w = 0; w < Wn; ++w) {
    float acc = bb;
#pragma unroll
    for (int ic = 0; ic < DCn; ++ic) {
      acc += ce[w + 0][ic] * wr[ic * 3 + 0];
      acc += ce[w + 1][ic] * wr[ic * 3 + 1];
      acc += ce[w + 2][ic] * wr[ic * 3 + 2];
    }
    m = fmaxf(m, acc);
  }
  xin[(size_t)bs * Dxn + DEn + oc] = (_Float16)m;
}

// ----------------------------- fused LSTM step -----------------------------
// One launch per scan step t (0..127); gridDim.z = direction. Block = 128
// threads (4 waves) computes a 32-row x 16-hidden-unit tile; wave g computes
// gate g's two 16x16 sub-tiles as one K=448 WMMA chain fusing the input
// projection (K<192) and the recurrence (K>=192). Each B fragment is reused
// by both sub-tiles.

__global__ __launch_bounds__(128) void lstm_step_kernel(
    const _Float16* __restrict__ xin,   // [128][512][192]
    const _Float16* __restrict__ wih,   // [2][1024][192] f16
    const _Float16* __restrict__ whh,   // [2][1024][256] f16
    const float*    __restrict__ bias,  // [2][1024]
    _Float16*       __restrict__ hbuf,  // [2 dir][2 parity][512][256] f16
    float*          __restrict__ cst,   // [2 dir][512][256]
    _Float16*       __restrict__ hall,  // [128][512][512] f16
    int t) {
  __shared__ _Float16 sA[32 * 448];     // 28 KB
  __shared__ float    sG[4][32][16];    // 8 KB

  const int n0  = blockIdx.x * 32;
  const int h0  = blockIdx.y * 16;
  const int dir = blockIdx.z;
  const int tx  = dir ? (Bn - 1 - t) : t;
  const int tid = threadIdx.x;

  const _Float16* hprev = hbuf + ((size_t)(dir * 2 + (t & 1))) * Sn * Hn;

  // vectorized cooperative A-tile fill: 32 rows x 448 K = 1792 x v8h
  {
    v8h* dst = (v8h*)sA;
    for (int idx = tid; idx < 32 * 56; idx += 128) {
      int r = idx / 56, c = idx % 56;
      v8h v;
      if (c < 24) v = ((const v8h*)(xin + ((size_t)tx * Sn + (n0 + r)) * Dxn))[c];
      else        v = ((const v8h*)(hprev + (size_t)(n0 + r) * Hn))[c - 24];
      dst[idx] = v;
    }
  }
  __syncthreads();

  const int gate = tid >> 5;
  const int lane = tid & 31;
  const int col  = lane & 15;
  const int hi   = lane >> 4;
  const int gcol = gate * Hn + h0 + col;

  const _Float16* wihd = wih + ((size_t)dir * Gn + gcol) * Dxn;
  const _Float16* whhd = whh + ((size_t)dir * Gn + gcol) * Hn;
  const int rowA = lane & 15;

  F32x8 acc0, acc1;
#pragma unroll
  for (int i = 0; i < 8; ++i) { acc0.e[i] = 0.0f; acc1.e[i] = 0.0f; }

#pragma unroll 2
  for (int kt = 0; kt < 14; ++kt) {
    const int k0 = kt * 32;
    // B fragment: lo-half K+0..7/+16..23, hi-half K+8..15/+24..31
    F16x16 bfr;
    const int kb = k0 + (hi ? 8 : 0);
    const _Float16* bp = (kt < 6) ? (wihd + kb) : (whhd + (kb - Dxn));
    __builtin_prefetch(bp + 32, 0, 1);
    bfr.h[0] = *(const v8h*)(bp);
    bfr.h[1] = *(const v8h*)(bp + 16);
    // A fragments (ISA 7.12.2): lo-half +0/+16, hi-half +8/+24; rows m0+lane%16
    F16x16 a0, a1;
    const _Float16* ap0 = sA + rowA * 448 + k0 + (hi ? 8 : 0);
    a0.h[0] = *(const v8h*)(ap0);
    a0.h[1] = *(const v8h*)(ap0 + 16);
    const _Float16* ap1 = ap0 + 16 * 448;
    a1.h[0] = *(const v8h*)(ap1);
    a1.h[1] = *(const v8h*)(ap1 + 16);
    acc0.v = __builtin_amdgcn_wmma_f32_16x16x32_f16(
        false, a0.v, false, bfr.v, (short)0, acc0.v, false, false);
    acc1.v = __builtin_amdgcn_wmma_f32_16x16x32_f16(
        false, a1.v, false, bfr.v, (short)0, acc1.v, false, false);
  }

  const float bb = bias[dir * Gn + gcol];
#pragma unroll
  for (int v = 0; v < 8; ++v) {
    sG[gate][v + hi * 8][col]      = acc0.e[v] + bb;   // D: M=v+8*hi, N=col
    sG[gate][16 + v + hi * 8][col] = acc1.e[v] + bb;
  }
  __syncthreads();

  _Float16* hnext = hbuf + ((size_t)(dir * 2 + ((t + 1) & 1))) * Sn * Hn;
  for (int e2 = tid; e2 < 32 * 16; e2 += 128) {
    int m = e2 >> 4, n = e2 & 15;
    float ig = sG[0][m][n], fg = sG[1][m][n];
    float gg = sG[2][m][n], og = sG[3][m][n];
    float i_ = 1.0f / (1.0f + __expf(-ig));
    float f_ = 1.0f / (1.0f + __expf(-fg));
    float g_ = tanhf(gg);
    float o_ = 1.0f / (1.0f + __expf(-og));
    int nrow = n0 + m, hu = h0 + n;
    float* cp = cst + ((size_t)dir * Sn + nrow) * Hn + hu;
    float c = f_ * (*cp) + i_ * g_;
    *cp = c;
    float h = o_ * tanhf(c);
    hnext[(size_t)nrow * Hn + hu] = (_Float16)h;
    hall[((size_t)tx * Sn + nrow) * H2n + dir * Hn + hu] = (_Float16)h;
  }
}

// ------------------------- FC + log_softmax (WMMA) -------------------------
// Block = 128 threads (4 waves) handles 32 rows of the flattened [B*S] axis.
// Wave w computes tag columns w*16..w*16+15 (fc_w zero-padded to 64 rows),
// K = 512 -> 16 WMMA k-steps, two M sub-tiles sharing each B fragment.

__global__ __launch_bounds__(128) void fc_logsoftmax_kernel(
    const _Float16* __restrict__ hall,  // [65536][512] f16
    const _Float16* __restrict__ fcw,   // [64][512] f16 (rows 50..63 zero)
    const float*    __restrict__ fcb,   // [50]
    float*          __restrict__ out) { // [65536][50]
  __shared__ _Float16 sA[32 * 512];     // 32 KB
  __shared__ float    sL[32][64];       // 8 KB
  __shared__ float    sLse[32];

  const int n0  = blockIdx.x * 32;
  const int tid = threadIdx.x;

  // 32 consecutive rows are contiguous in hall -> linear vector copy
  {
    const v8h* src = (const v8h*)(hall + (size_t)n0 * H2n);
    v8h* dst = (v8h*)sA;
    for (int i = tid; i < 32 * 512 / 8; i += 128) dst[i] = src[i];
  }
  __syncthreads();

  const int wv   = tid >> 5;
  const int lane = tid & 31;
  const int col  = lane & 15;
  const int hi   = lane >> 4;
  const int tag  = wv * 16 + col;
  const int rowA = lane & 15;

  F32x8 acc0, acc1;
#pragma unroll
  for (int i = 0; i < 8; ++i) { acc0.e[i] = 0.0f; acc1.e[i] = 0.0f; }

#pragma unroll 2
  for (int kt = 0; kt < 16; ++kt) {
    const int k0 = kt * 32;
    F16x16 bfr;
    const _Float16* bp = fcw + (size_t)tag * 512 + k0 + (hi ? 8 : 0);
    bfr.h[0] = *(const v8h*)(bp);
    bfr.h[1] = *(const v8h*)(bp + 16);
    F16x16 a0, a1;
    const _Float16* ap0 = sA + rowA * 512 + k0 + (hi ? 8 : 0);
    a0.h[0] = *(const v8h*)(ap0);
    a0.h[1] = *(const v8h*)(ap0 + 16);
    const _Float16* ap1 = ap0 + 16 * 512;
    a1.h[0] = *(const v8h*)(ap1);
    a1.h[1] = *(const v8h*)(ap1 + 16);
    acc0.v = __builtin_amdgcn_wmma_f32_16x16x32_f16(
        false, a0.v, false, bfr.v, (short)0, acc0.v, false, false);
    acc1.v = __builtin_amdgcn_wmma_f32_16x16x32_f16(
        false, a1.v, false, bfr.v, (short)0, acc1.v, false, false);
  }

  const float bb = (tag < Tn) ? fcb[tag] : 0.0f;
#pragma unroll
  for (int v = 0; v < 8; ++v) {
    sL[v + hi * 8][wv * 16 + col]      = acc0.e[v] + bb;
    sL[16 + v + hi * 8][wv * 16 + col] = acc1.e[v] + bb;
  }
  __syncthreads();

  if (tid < 32) {
    float mx = -3.4e38f;
    for (int c = 0; c < Tn; ++c) mx = fmaxf(mx, sL[tid][c]);
    float se = 0.0f;
    for (int c = 0; c < Tn; ++c) se += __expf(sL[tid][c] - mx);
    sLse[tid] = mx + __logf(se);
  }
  __syncthreads();

  for (int idx = tid; idx < 32 * Tn; idx += 128) {
    int m = idx / Tn, c = idx % Tn;
    out[(size_t)(n0 + m) * Tn + c] = sL[m][c] - sLse[m];
  }
}

// ------------------------------- launcher ----------------------------------

extern "C" void kernel_launch(void* const* d_in, const int* in_sizes, int n_in,
                              void* d_out, int out_size, void* d_ws, size_t ws_size,
                              hipStream_t stream) {
  const int*   words = (const int*)d_in[0];
  const int*   chars = (const int*)d_in[1];
  const int*   wlens = (const int*)d_in[2];
  const int*   clens = (const int*)d_in[3];
  const float* wemb  = (const float*)d_in[4];
  const float* cemb  = (const float*)d_in[5];
  const float* convw = (const float*)d_in[6];
  const float* convb = (const float*)d_in[7];
  const float* wihf  = (const float*)d_in[8];
  const float* whhf  = (const float*)d_in[9];
  const float* bihf  = (const float*)d_in[10];
  const float* bhhf  = (const float*)d_in[11];
  const float* wihb  = (const float*)d_in[12];
  const float* whhb  = (const float*)d_in[13];
  const float* bihb  = (const float*)d_in[14];
  const float* bhhb  = (const float*)d_in[15];
  const float* fcwf  = (const float*)d_in[16];
  const float* fcbf  = (const float*)d_in[17];
  float* out = (float*)d_out;
  (void)in_sizes; (void)n_in; (void)out_size; (void)ws_size;

  // workspace layout (region sizes all multiples of 256 B; total ~96 MB)
  char* ws = (char*)d_ws;
  size_t off = 0;
  _Float16* xin  = (_Float16*)(ws + off); off += (size_t)NBS * Dxn * 2;
  _Float16* wih  = (_Float16*)(ws + off); off += (size_t)2 * Gn * Dxn * 2;
  _Float16* whh  = (_Float16*)(ws + off); off += (size_t)2 * Gn * Hn * 2;
  _Float16* fcw  = (_Float16*)(ws + off); off += (size_t)64 * H2n * 2;
  float*    bias = (float*)   (ws + off); off += (size_t)2 * Gn * 4;
  _Float16* hbuf = (_Float16*)(ws + off); off += (size_t)2 * 2 * Sn * Hn * 2;
  float*    cst  = (float*)   (ws + off); off += (size_t)2 * Sn * Hn * 4;
  _Float16* hall = (_Float16*)(ws + off); off += (size_t)NBS * H2n * 2;

  auto cvt = [&](const float* s, _Float16* d, int n) {
    cvt_f32_f16_kernel<<<(n + 255) / 256, 256, 0, stream>>>(s, d, n);
  };

  cvt(wihf, wih,            Gn * Dxn);
  cvt(wihb, wih + Gn * Dxn, Gn * Dxn);
  cvt(whhf, whh,            Gn * Hn);
  cvt(whhb, whh + Gn * Hn,  Gn * Hn);
  zero_f16_kernel<<<(64 * H2n + 255) / 256, 256, 0, stream>>>(fcw, 64 * H2n);
  cvt(fcwf, fcw, Tn * H2n);
  bias_sum_kernel<<<(2 * Gn + 255) / 256, 256, 0, stream>>>(bihf, bhhf, bihb, bhhb, bias);
  zero_f16_kernel<<<(2 * 2 * Sn * Hn + 255) / 256, 256, 0, stream>>>(hbuf, 2 * 2 * Sn * Hn);
  zero_f32_kernel<<<(2 * Sn * Hn + 255) / 256, 256, 0, stream>>>(cst, 2 * Sn * Hn);

  embed_word_kernel<<<NBS, 128, 0, stream>>>(words, wlens, wemb, xin);
  char_cnn_kernel<<<NBS, 64, 0, stream>>>(chars, wlens, clens, cemb, convw, convb, xin);

  for (int t = 0; t < Bn; ++t)
    lstm_step_kernel<<<dim3(Sn / 32, Hn / 16, 2), 128, 0, stream>>>(
        xin, wih, whh, bias, hbuf, cst, hall, t);

  fc_logsoftmax_kernel<<<NBS / 32, 128, 0, stream>>>(hall, fcw, fcbf, out);
}
